// MultiHeadAttention_81767587381248
// MI455X (gfx1250) — matrix operand order repaired and checked by reading, not compiled, not run on hardware
//
#include <hip/hip_runtime.h>
#include <hip/hip_bf16.h>

typedef __attribute__((ext_vector_type(16))) __bf16 v16bf;
typedef __attribute__((ext_vector_type(8)))  float  v8f;

// f32 -> bf16 round-to-nearest-even
static __device__ inline __bf16 f2bf(float f) {
    union { float f; unsigned u; } x; x.f = f;
    unsigned r = (x.u + 0x7FFFu + ((x.u >> 16) & 1u)) >> 16;
    union { unsigned short s; __bf16 b; } y; y.s = (unsigned short)r;
    return y.b;
}

// ---- CDNA5 async global->LDS copy (32 bytes per thread) + wait --------------
// LDS destination address is the first vgpr operand, global source the second; both add the offset
static __device__ inline unsigned lds_off(const void* p) {
    // generic shared pointer: ADDR[31:0] is the wave-relative LDS address
    return (unsigned)(size_t)p;
}
static __device__ inline void async_copy32B(const void* gsrc, void* ldst) {
    unsigned l = lds_off(ldst);
    asm volatile(
        "global_load_async_to_lds_b128 %0, %1, off\n\t"
        "global_load_async_to_lds_b128 %0, %1, off offset:16"
        :: "v"(l), "v"(gsrc) : "memory");
}
static __device__ inline void wait_asynccnt0() {
    asm volatile("s_wait_asynccnt 0x0" ::: "memory");
}

// ---- f32 -> bf16 conversion pass -------------------------------------------
__global__ __launch_bounds__(256) void cvt_f32_bf16(
    const float* __restrict__ src, __bf16* __restrict__ dst, int n)
{
    int i = (blockIdx.x * 256 + threadIdx.x) * 4;
    if (i < n) {
        float4 f = *(const float4*)(src + i);
        dst[i + 0] = f2bf(f.x);
        dst[i + 1] = f2bf(f.y);
        dst[i + 2] = f2bf(f.z);
        dst[i + 3] = f2bf(f.w);
    }
}

// ---- WMMA GEMM: C = A @ W.T, A [M,K] bf16, W [N,K] bf16 --------------------
#define BM 128
#define BN 128
#define BK 32
#define LDT 40   // BK + 8 pad: 80B row stride -> 16B-aligned b128 dsts, 20-bank stride

template <bool OUT_BF16>
__global__ __launch_bounds__(256) void gemm_wmma_async(
    const __bf16* __restrict__ Ap,
    const __bf16* __restrict__ B0, const __bf16* __restrict__ B1, const __bf16* __restrict__ B2,
    void* __restrict__ O0, void* __restrict__ O1, void* __restrict__ O2,
    int M, int N, int K)
{
    // double-buffered tiles: [buf][A 128x40 | B 128x40]
    __shared__ __bf16 smem[2][(BM + BN) * LDT];

    const __bf16* Bw = (blockIdx.z == 0) ? B0 : ((blockIdx.z == 1) ? B1 : B2);
    void*         Op = (blockIdx.z == 0) ? O0 : ((blockIdx.z == 1) ? O1 : O2);

    const int tid     = threadIdx.x;
    const int lane    = tid & 31;
    const int wave    = tid >> 5;
    const int wm      = wave & 3;   // 4 wave rows  -> 32 output rows each
    const int wn      = wave >> 2;  // 2 wave cols  -> 64 output cols each
    const int lr      = lane & 15;
    const int halfsel = lane >> 4;

    const int bm = blockIdx.y * BM;
    const int bn = blockIdx.x * BN;

    // per-thread tile-load slice: 2 threads/row, 16 bf16 (32B) each
    const int lrow = tid >> 1;
    const int lcol = (tid & 1) * 16;
    const __bf16* gA = Ap + (size_t)(bm + lrow) * K + lcol;
    const __bf16* gB = Bw + (size_t)(bn + lrow) * K + lcol;

    v8f acc[2][4] = {};

    const int niter = K / BK;
    // prefetch tile 0 into buffer 0
    async_copy32B(gA, &smem[0][lrow * LDT + lcol]);
    async_copy32B(gB, &smem[0][BM * LDT + lrow * LDT + lcol]);

    for (int it = 0; it < niter; ++it) {
        const int cur = it & 1;
        wait_asynccnt0();
        __syncthreads();   // tile[cur] visible to all; all reads of tile[cur^1] retired

        if (it + 1 < niter) {
            int kb = (it + 1) * BK;
            async_copy32B(gA + kb, &smem[cur ^ 1][lrow * LDT + lcol]);
            async_copy32B(gB + kb, &smem[cur ^ 1][BM * LDT + lrow * LDT + lcol]);
        }

        const __bf16* sA = &smem[cur][0];
        const __bf16* sB = &smem[cur][BM * LDT];

        // fragments per ISA 16-bit layouts
        v16bf afr[2], bfr[4];
#pragma unroll
        for (int tm = 0; tm < 2; ++tm) {
            const __bf16* rp = sA + (wm * 32 + tm * 16 + lr) * LDT;
#pragma unroll
            for (int j = 0; j < 16; ++j) {
                int k = j + ((j >= 8) ? 8 : 0) + halfsel * 8;   // A 16x32
                afr[tm][j] = rp[k];
            }
        }
#pragma unroll
        for (int tn = 0; tn < 4; ++tn) {
            const __bf16* cp = sB + (wn * 64 + tn * 16 + lr) * LDT;
#pragma unroll
            for (int j = 0; j < 16; ++j) {
                int k = j + ((j >= 8) ? 8 : 0) + halfsel * 8;   // B 32x16
                bfr[tn][j] = cp[k];
            }
        }
#pragma unroll
        for (int tm = 0; tm < 2; ++tm)
#pragma unroll
            for (int tn = 0; tn < 4; ++tn)
                acc[tm][tn] = __builtin_amdgcn_wmma_f32_16x16x32_bf16(
                    false, afr[tm], false, bfr[tn], (short)0, acc[tm][tn], false, false);
    }

    // epilogue: C layout lane->N, vgpr v -> M = v + 8*halfsel
#pragma unroll
    for (int tm = 0; tm < 2; ++tm) {
#pragma unroll
        for (int tn = 0; tn < 4; ++tn) {
            int n = bn + wn * 64 + tn * 16 + lr;
#pragma unroll
            for (int v = 0; v < 8; ++v) {
                int m = bm + wm * 32 + tm * 16 + v + halfsel * 8;
                if constexpr (OUT_BF16)
                    ((__bf16*)Op)[(size_t)m * N + n] = f2bf(acc[tm][tn][v]);
                else
                    ((float*)Op)[(size_t)m * N + n] = acc[tm][tn][v];
            }
        }
    }
}

// ---- per-token attention over the heads axis (faithful to reference) -------
// scores = Q[t] @ K[t]^T / 8 (16x64 @ 64x16), softmax over last axis,
// out = attn @ V[t] (16x16 @ 16x64). One wave per token, 8 tokens/block.
__global__ __launch_bounds__(256) void attn_wmma(
    const __bf16* __restrict__ Q, const __bf16* __restrict__ Km,
    const __bf16* __restrict__ V, __bf16* __restrict__ O)
{
    __shared__ float sAttn[8][16][17];

    const int tid     = threadIdx.x;
    const int lane    = tid & 31;
    const int wave    = tid >> 5;
    const int lr      = lane & 15;
    const int halfsel = lane >> 4;

    const size_t t = (size_t)blockIdx.x * 8 + wave;
    const __bf16* q = Q  + t * 1024;
    const __bf16* k = Km + t * 1024;
    const __bf16* v = V  + t * 1024;

    union PK { unsigned u[8]; __bf16 b[16]; };

    v16bf qa[2], kb[2];
    {
        const unsigned* qp = (const unsigned*)(q + (size_t)lr * 64);
#pragma unroll
        for (int f = 0; f < 2; ++f) {
            PK p;
#pragma unroll
            for (int jj = 0; jj < 8; ++jj) {
                int kk = f * 32 + halfsel * 8 + 2 * jj + ((jj >= 4) ? 8 : 0);
                p.u[jj] = qp[kk >> 1];
            }
#pragma unroll
            for (int j = 0; j < 16; ++j) qa[f][j] = p.b[j];
        }
    }
    {
        const unsigned* kp = (const unsigned*)(k + (size_t)lr * 64);
#pragma unroll
        for (int f = 0; f < 2; ++f) {
            PK p;
#pragma unroll
            for (int jj = 0; jj < 8; ++jj)
                p.u[jj] = kp[f * 16 + halfsel * 4 + jj + ((jj >= 4) ? 4 : 0)];
#pragma unroll
            for (int j = 0; j < 16; ++j) kb[f][j] = p.b[j];
        }
    }

    v8f sc = {};
    sc = __builtin_amdgcn_wmma_f32_16x16x32_bf16(false, qa[0], false, kb[0], (short)0, sc, false, false);
    sc = __builtin_amdgcn_wmma_f32_16x16x32_bf16(false, qa[1], false, kb[1], (short)0, sc, false, false);

    const float scale = 0.125f;  // 1/sqrt(64)
#pragma unroll
    for (int e = 0; e < 8; ++e) {
        float x0 = sc[e] * scale;
        float mx = x0;
#pragma unroll
        for (int off = 1; off < 16; off <<= 1)
            mx = fmaxf(mx, __shfl_xor(mx, off, 32));
        float ex = __expf(x0 - mx);
        float sum = ex;
#pragma unroll
        for (int off = 1; off < 16; off <<= 1)
            sum += __shfl_xor(sum, off, 32);
        float pr = ex / sum;
        int m = e + halfsel * 8;
        sAttn[wave][m][lr] = pr;
    }
    __syncthreads();

    v16bf afr;
#pragma unroll
    for (int j = 0; j < 16; ++j)
        afr[j] = (halfsel == 0) ? f2bf(sAttn[wave][lr][j]) : f2bf(0.0f);

#pragma unroll
    for (int d0 = 0; d0 < 64; d0 += 16) {
        v16bf vfr;
#pragma unroll
        for (int j = 0; j < 16; ++j) {
            __bf16 val = v[(size_t)j * 64 + d0 + lr];
            vfr[j] = halfsel ? f2bf(0.0f) : val;
        }
        v8f o = {};
        o = __builtin_amdgcn_wmma_f32_16x16x32_bf16(false, afr, false, vfr, (short)0, o, false, false);
#pragma unroll
        for (int e = 0; e < 8; ++e) {
            int m = e + halfsel * 8;
            O[t * 1024 + (size_t)m * 64 + d0 + lr] = f2bf(o[e]);
        }
    }
}

extern "C" void kernel_launch(void* const* d_in, const int* in_sizes, int n_in,
                              void* d_out, int out_size, void* d_ws, size_t ws_size,
                              hipStream_t stream) {
    const float* x   = (const float*)d_in[0];
    const float* Wq  = (const float*)d_in[1];
    const float* Wk  = (const float*)d_in[2];
    const float* Wv  = (const float*)d_in[3];
    const float* Wfc = (const float*)d_in[4];

    const int DM = 1024;
    const int MT = in_sizes[0] / DM;           // B*S = 16384
    const size_t elems = (size_t)MT * DM;      // 16.7M
    const size_t welem = (size_t)DM * DM;      // 1M

    __bf16* Qb  = (__bf16*)d_ws;
    __bf16* Kb  = Qb  + elems;
    __bf16* Vb  = Kb  + elems;
    __bf16* AO  = Vb  + elems;
    __bf16* Xb  = AO  + elems;
    __bf16* Wqb = Xb  + elems;
    __bf16* Wkb = Wqb + welem;
    __bf16* Wvb = Wkb + welem;
    __bf16* Wfb = Wvb + welem;

    // 1) convert all f32 operands to bf16 once
    cvt_f32_bf16<<<(int)(elems / 1024), 256, 0, stream>>>(x,   Xb,  (int)elems);
    cvt_f32_bf16<<<(int)(welem / 1024), 256, 0, stream>>>(Wq,  Wqb, (int)welem);
    cvt_f32_bf16<<<(int)(welem / 1024), 256, 0, stream>>>(Wk,  Wkb, (int)welem);
    cvt_f32_bf16<<<(int)(welem / 1024), 256, 0, stream>>>(Wv,  Wvb, (int)welem);
    cvt_f32_bf16<<<(int)(welem / 1024), 256, 0, stream>>>(Wfc, Wfb, (int)welem);

    // 2) Q/K/V = x @ W{q,k,v}.T   (z selects weight/output; async double-buffered tiles)
    gemm_wmma_async<true><<<dim3(DM / BN, MT / BM, 3), 256, 0, stream>>>(
        Xb, Wqb, Wkb, Wvb, (void*)Qb, (void*)Kb, (void*)Vb, MT, DM, DM);

    // 3) per-token attention over heads axis
    attn_wmma<<<MT / 8, 256, 0, stream>>>(Qb, Kb, Vb, AO);

    // 4) out = attn_out @ Wfc.T  (f32 straight to d_out)
    gemm_wmma_async<false><<<dim3(DM / BN, MT / BM, 1), 256, 0, stream>>>(
        AO, Wfb, Wfb, Wfb, d_out, d_out, d_out, MT, DM, DM);
}
